// ExampleNet_9594956939892
// MI455X (gfx1250) — matrix operand order repaired and checked by reading, not compiled, not run on hardware
//
#include <hip/hip_runtime.h>

typedef __attribute__((ext_vector_type(16))) _Float16 v16h;
typedef __attribute__((ext_vector_type(8)))  float    v8f;

#define N_NODES  50000
#define N_EDGES  400000
#define N_GRAPHS 2000
#define NF 11
#define EF 4

// conv1: K = 352 (+11 bias rows, pad to 384) -> 12 k-steps of 32, N = 32 (2 tiles)
// conv2: K = 1024 (+32 bias rows)           -> 33 k-steps of 32, N = 16 (1 tile)
#define K1STEPS 12
#define K2STEPS 33
#define N_TILES (N_EDGES / 16)   // 25000 exactly

// ---------------------------------------------------------------- utilities

__global__ void zero_f32(float* __restrict__ p, int n) {
  int i = blockIdx.x * blockDim.x + threadIdx.x;
  if (i < n) p[i] = 0.0f;
}

// Build B matrices pre-swizzled into per-lane WMMA fragment order:
//   layout [kstep][ntile][lane][16 halves]  (each lane loads one contiguous v16h)
// B fragment (16-bit, Kx16): lane l holds col N = l&15; element j holds
//   K = kstep*32 + (j>=8 ? 16 : 0) + (l>=16 ? 8 : 0) + (j&7).
// conv1: Br1[K][N] = w2_1[(K&31)][ (K>>5)*32 + N ]   for K<352  (z flatten j=i*32+k)
//                  = b2_1[(K-352)*32 + N]            for 352<=K<363, else 0
// conv2: Br2[K][N] = w2_2[(K&31)][ (K>>5)*16 + N ]   for K<1024
//                  = b2_2[(K-1024)*16 + N]           for 1024<=K<1056
__global__ void build_bfrag(const float* __restrict__ w2_1, const float* __restrict__ b2_1,
                            const float* __restrict__ w2_2, const float* __restrict__ b2_2,
                            _Float16* __restrict__ bf1, _Float16* __restrict__ bf2) {
  int t = blockIdx.x * blockDim.x + threadIdx.x;
  const int n1 = K1STEPS * 2 * 32 * 16;
  const int n2 = K2STEPS * 32 * 16;
  if (t < n1) {
    int j = t & 15, lane = (t >> 4) & 31, nt = (t >> 9) & 1, ks = t >> 10;
    int N = nt * 16 + (lane & 15);
    int K = ks * 32 + ((j >> 3) << 4) + ((lane >> 4) << 3) + (j & 7);
    float v = 0.0f;
    if (K < 352)      v = w2_1[(K & 31) * 352 + (K >> 5) * 32 + N];
    else if (K < 363) v = b2_1[(K - 352) * 32 + N];
    bf1[t] = (_Float16)v;
  } else if (t < n1 + n2) {
    int u = t - n1;
    int j = u & 15, lane = (u >> 4) & 31, ks = u >> 9;
    int N = lane & 15;
    int K = ks * 32 + ((j >> 3) << 4) + ((lane >> 4) << 3) + (j & 7);
    float v = 0.0f;
    if (K < 1024)      v = w2_2[(K & 31) * 512 + (K >> 5) * 16 + N];
    else if (K < 1056) v = b2_2[(K - 1024) * 16 + N];
    bf2[u] = (_Float16)v;
  }
}

// ------------------------------------------------- conv1 fused edge kernel
// One wave per 16-edge tile. msg[16,32] = Z[16,384] @ Br1[384,32] via WMMA.
__global__ void __launch_bounds__(256)
conv1_msg(const float* __restrict__ x, const float* __restrict__ edge_attr,
          const float* __restrict__ w1, const float* __restrict__ b1,
          const int* __restrict__ eidx, const _Float16* __restrict__ bfrag,
          float* __restrict__ agg) {
  __shared__ float sW[EF * 32 + 32];
  for (int i = threadIdx.x; i < EF * 32 + 32; i += blockDim.x)
    sW[i] = (i < EF * 32) ? w1[i] : b1[i - EF * 32];
  __syncthreads();

  const int lane = threadIdx.x & 31;
  const int wv   = threadIdx.x >> 5;
  const int m    = lane & 15;
  const bool hi  = lane >= 16;
  const int tile = blockIdx.x * 8 + wv;
  if (tile >= N_TILES) return;                 // wave-uniform branch

  const int e   = tile * 16 + m;
  const int src = eidx[e];

  _Float16 xh[NF];
  #pragma unroll
  for (int i = 0; i < NF; ++i) xh[i] = (_Float16)x[src * NF + i];

  const float ea0 = edge_attr[e * 4 + 0], ea1 = edge_attr[e * 4 + 1];
  const float ea2 = edge_attr[e * 4 + 2], ea3 = edge_attr[e * 4 + 3];
  _Float16 hv[32];
  #pragma unroll
  for (int c = 0; c < 32; ++c) {
    float v = sW[128 + c] + ea0 * sW[c] + ea1 * sW[32 + c] + ea2 * sW[64 + c] + ea3 * sW[96 + c];
    hv[c] = (_Float16)fmaxf(v, 0.0f);
  }

  // A-fragment (16x32 f16) of the edge-MLP hidden vector h
  v16h hfrag;
  #pragma unroll
  for (int j = 0; j < 16; ++j) {
    const int ia = ((j >> 3) << 4) | (j & 7);
    hfrag[j] = hi ? hv[ia + 8] : hv[ia];
  }

  const v16h* Bf = (const v16h*)bfrag;
  v8f acc0 = {0,0,0,0,0,0,0,0}, acc1 = {0,0,0,0,0,0,0,0};

  #pragma unroll
  for (int kb = 0; kb < 11; ++kb) {
    const _Float16 s = xh[kb];
    v16h a;
    #pragma unroll
    for (int j = 0; j < 16; ++j) a[j] = (_Float16)(hfrag[j] * s);
    v16h b0v = Bf[(kb * 2 + 0) * 32 + lane];
    v16h b1v = Bf[(kb * 2 + 1) * 32 + lane];
    acc0 = __builtin_amdgcn_wmma_f32_16x16x32_f16(false, a, false, b0v, (short)0, acc0, false, false);
    acc1 = __builtin_amdgcn_wmma_f32_16x16x32_f16(false, a, false, b1v, (short)0, acc1, false, false);
  }
  { // bias K-block (folds b2_1 into the GEMM): z = [x_src ; 0]
    v16h a;
    #pragma unroll
    for (int j = 0; j < 16; ++j) {
      const int ia = ((j >> 3) << 4) | (j & 7);
      const _Float16 va = (ia     < NF) ? xh[ia]     : (_Float16)0.0f;
      const _Float16 vb = (ia + 8 < NF) ? xh[ia + 8] : (_Float16)0.0f;
      a[j] = hi ? vb : va;
    }
    v16h b0v = Bf[(11 * 2 + 0) * 32 + lane];
    v16h b1v = Bf[(11 * 2 + 1) * 32 + lane];
    acc0 = __builtin_amdgcn_wmma_f32_16x16x32_f16(false, a, false, b0v, (short)0, acc0, false, false);
    acc1 = __builtin_amdgcn_wmma_f32_16x16x32_f16(false, a, false, b1v, (short)0, acc1, false, false);
  }

  // D layout: VGPR r, lanes 0-15 -> row r, lanes 16-31 -> row 8+r; col = lane&15
  const int rowbase = tile * 16 + (hi ? 8 : 0);
  #pragma unroll
  for (int r = 0; r < 8; ++r) {
    const int dst = eidx[N_EDGES + rowbase + r];
    atomicAdd(&agg[dst * 32 + m],      acc0[r]);
    atomicAdd(&agg[dst * 32 + 16 + m], acc1[r]);
  }
}

// ------------------------------------------------- conv2 fused edge kernel
__global__ void __launch_bounds__(256)
conv2_msg(const float* __restrict__ h1, const float* __restrict__ edge_attr,
          const float* __restrict__ w1, const float* __restrict__ b1,
          const int* __restrict__ eidx, const _Float16* __restrict__ bfrag,
          float* __restrict__ agg) {
  __shared__ float sW[EF * 32 + 32];
  for (int i = threadIdx.x; i < EF * 32 + 32; i += blockDim.x)
    sW[i] = (i < EF * 32) ? w1[i] : b1[i - EF * 32];
  __syncthreads();

  const int lane = threadIdx.x & 31;
  const int wv   = threadIdx.x >> 5;
  const int m    = lane & 15;
  const bool hi  = lane >= 16;
  const int tile = blockIdx.x * 8 + wv;
  if (tile >= N_TILES) return;

  const int e   = tile * 16 + m;
  const int src = eidx[e];

  _Float16 hs[32];
  #pragma unroll
  for (int i = 0; i < 32; ++i) hs[i] = (_Float16)h1[src * 32 + i];

  const float ea0 = edge_attr[e * 4 + 0], ea1 = edge_attr[e * 4 + 1];
  const float ea2 = edge_attr[e * 4 + 2], ea3 = edge_attr[e * 4 + 3];
  _Float16 hv[32];
  #pragma unroll
  for (int c = 0; c < 32; ++c) {
    float v = sW[128 + c] + ea0 * sW[c] + ea1 * sW[32 + c] + ea2 * sW[64 + c] + ea3 * sW[96 + c];
    hv[c] = (_Float16)fmaxf(v, 0.0f);
  }

  v16h h2frag, hsfrag;
  #pragma unroll
  for (int j = 0; j < 16; ++j) {
    const int ia = ((j >> 3) << 4) | (j & 7);
    h2frag[j] = hi ? hv[ia + 8] : hv[ia];
    hsfrag[j] = hi ? hs[ia + 8] : hs[ia];
  }

  const v16h* Bf = (const v16h*)bfrag;
  v8f acc = {0,0,0,0,0,0,0,0};

  #pragma unroll
  for (int kb = 0; kb < 32; ++kb) {
    const _Float16 s = hs[kb];
    v16h a;
    #pragma unroll
    for (int j = 0; j < 16; ++j) a[j] = (_Float16)(h2frag[j] * s);
    v16h bv = Bf[kb * 32 + lane];
    acc = __builtin_amdgcn_wmma_f32_16x16x32_f16(false, a, false, bv, (short)0, acc, false, false);
  }
  { // bias K-block: z = h1[src] itself
    v16h bv = Bf[32 * 32 + lane];
    acc = __builtin_amdgcn_wmma_f32_16x16x32_f16(false, hsfrag, false, bv, (short)0, acc, false, false);
  }

  const int rowbase = tile * 16 + (hi ? 8 : 0);
  #pragma unroll
  for (int r = 0; r < 8; ++r) {
    const int dst = eidx[N_EDGES + rowbase + r];
    atomicAdd(&agg[dst * 16 + m], acc[r]);
  }
}

// --------------------------------------------------------- node + head ops

__global__ void node1(const float* __restrict__ agg, const float* __restrict__ x,
                      const float* __restrict__ root, const float* __restrict__ bias,
                      float* __restrict__ h1) {
  int t = blockIdx.x * blockDim.x + threadIdx.x;
  if (t >= N_NODES * 32) return;
  const int n = t >> 5, c = t & 31;
  float v = agg[t] + bias[c];
  #pragma unroll
  for (int i = 0; i < NF; ++i) v += x[n * NF + i] * root[i * 32 + c];
  h1[t] = fmaxf(v, 0.0f);
}

__global__ void node2(const float* __restrict__ agg, const float* __restrict__ h1,
                      const float* __restrict__ root, const float* __restrict__ bias,
                      const int* __restrict__ batch, float* __restrict__ g) {
  int t = blockIdx.x * blockDim.x + threadIdx.x;
  if (t >= N_NODES * 16) return;
  const int n = t >> 4, c = t & 15;
  float v = agg[t] + bias[c];
  #pragma unroll
  for (int i = 0; i < 32; ++i) v += h1[n * 32 + i] * root[i * 16 + c];
  atomicAdd(&g[batch[n] * 16 + c], fmaxf(v, 0.0f));
}

__global__ void head(const float* __restrict__ g, const float* __restrict__ fc1w,
                     const float* __restrict__ fc1b, const float* __restrict__ outw,
                     const float* __restrict__ outb, float* __restrict__ out) {
  int t = blockIdx.x * blockDim.x + threadIdx.x;
  if (t >= N_GRAPHS) return;
  float gg[16];
  #pragma unroll
  for (int i = 0; i < 16; ++i) gg[i] = g[t * 16 + i];
  float o = outb[0];
  #pragma unroll
  for (int c = 0; c < 32; ++c) {
    float v = fc1b[c];
    #pragma unroll
    for (int i = 0; i < 16; ++i) v += gg[i] * fc1w[i * 32 + c];
    o += fmaxf(v, 0.0f) * outw[c];
  }
  out[t] = o;
}

// ------------------------------------------------------------------ launch

extern "C" void kernel_launch(void* const* d_in, const int* in_sizes, int n_in,
                              void* d_out, int out_size, void* d_ws, size_t ws_size,
                              hipStream_t stream) {
  (void)in_sizes; (void)n_in; (void)out_size; (void)ws_size;
  const float* x     = (const float*)d_in[0];
  const float* ea    = (const float*)d_in[1];
  const float* w1_1  = (const float*)d_in[2];
  const float* b1_1  = (const float*)d_in[3];
  const float* w2_1  = (const float*)d_in[4];
  const float* b2_1  = (const float*)d_in[5];
  const float* root1 = (const float*)d_in[6];
  const float* bias1 = (const float*)d_in[7];
  const float* w1_2  = (const float*)d_in[8];
  const float* b1_2  = (const float*)d_in[9];
  const float* w2_2  = (const float*)d_in[10];
  const float* b2_2  = (const float*)d_in[11];
  const float* root2 = (const float*)d_in[12];
  const float* bias2 = (const float*)d_in[13];
  const float* fc1w  = (const float*)d_in[14];
  const float* fc1b  = (const float*)d_in[15];
  const float* outw  = (const float*)d_in[16];
  const float* outb  = (const float*)d_in[17];
  const int*   eidx  = (const int*)d_in[18];
  const int*   batch = (const int*)d_in[19];
  float* out = (float*)d_out;

  // workspace layout (16.2 MB): [agg1][agg2][g][h1][bfrag1][bfrag2]
  float* agg1 = (float*)d_ws;               // N_NODES*32
  float* agg2 = agg1 + N_NODES * 32;        // N_NODES*16
  float* gbuf = agg2 + N_NODES * 16;        // N_GRAPHS*16
  float* h1   = gbuf + N_GRAPHS * 16;       // N_NODES*32
  _Float16* bf1 = (_Float16*)(h1 + N_NODES * 32);        // 12*2*32*16
  _Float16* bf2 = bf1 + K1STEPS * 2 * 32 * 16;           // 33*32*16

  const int nz = N_NODES * 32 + N_NODES * 16 + N_GRAPHS * 16;  // contiguous zero region
  zero_f32<<<(nz + 255) / 256, 256, 0, stream>>>(agg1, nz);

  const int nb = K1STEPS * 2 * 32 * 16 + K2STEPS * 32 * 16;
  build_bfrag<<<(nb + 255) / 256, 256, 0, stream>>>(w2_1, b2_1, w2_2, b2_2, bf1, bf2);

  conv1_msg<<<N_TILES / 8, 256, 0, stream>>>(x, ea, w1_1, b1_1, eidx, bf1, agg1);
  node1<<<(N_NODES * 32 + 255) / 256, 256, 0, stream>>>(agg1, x, root1, bias1, h1);
  conv2_msg<<<N_TILES / 8, 256, 0, stream>>>(h1, ea, w1_2, b1_2, eidx, bf2, agg2);
  node2<<<(N_NODES * 16 + 255) / 256, 256, 0, stream>>>(agg2, h1, root2, bias2, batch, gbuf);
  head<<<(N_GRAPHS + 255) / 256, 256, 0, stream>>>(gbuf, fc1w, fc1b, outw, outb, out);
}
